// Skinning_81277961109692
// MI455X (gfx1250) — matrix operand order repaired and checked by reading, not compiled, not run on hardware
//
#include <hip/hip_runtime.h>

typedef __attribute__((ext_vector_type(16))) _Float16 v16h;
typedef __attribute__((ext_vector_type(8)))  _Float16 v8h;
typedef __attribute__((ext_vector_type(8)))  float    v8f;
typedef __attribute__((ext_vector_type(4)))  float    v4f;

#define NJ 24

__device__ __constant__ int d_par[NJ] = {-1,0,0,0,1,2,3,4,5,6,7,8,9,9,9,12,13,14,16,17,18,19,20,21};

// ---------------------------------------------------------------------------
// Prep: build per-batch 32x32 f16 blend tables in d_ws.
//   rows 0..11  : A1_inv[n][r][c]   (star pose, batch independent)
//   rows 12..23 : A2[b][n][r][c]
//   row  24     : 1.0 (produces W = sum of weights)
//   all other rows/cols: 0
// One wave: lane b<16 walks chain for batch b, lane 16 walks the star chain.
// ---------------------------------------------------------------------------
__global__ void skin_prep(const float* __restrict__ pose,
                          const float* __restrict__ scale,
                          const float* __restrict__ trans,
                          const float* __restrict__ Jin,
                          _Float16* __restrict__ tab, int B)
{
    __shared__ float GLS[17][NJ][12];   // chained global transforms (3x4)
    __shared__ float ALS[17][NJ][12];   // relative transforms A (3x4)
    __shared__ float AINV[NJ][12];      // inverse of star-pose A (3x4)

    const int lane = threadIdx.x;
    const float s   = scale[0];
    const float tr0 = trans[0], tr1 = trans[1], tr2 = trans[2];

    if (lane < 17) {
        for (int i = 0; i < NJ; ++i) {
            float px, py, pz;
            if (lane < 16) {
                const float* pp = pose + lane * 72 + i * 3;
                px = pp[0]; py = pp[1]; pz = pp[2];
            } else {                       // star pose: pose[5]=0.5, pose[8]=-0.5
                px = 0.f; py = 0.f; pz = 0.f;
                if (i == 1) pz = 0.5f;
                if (i == 2) pz = -0.5f;
            }
            // Rodrigues
            float th = sqrtf(px*px + py*py + pz*pz) + 1e-8f;
            float ux = px/th, uy = py/th, uz = pz/th;
            float c = cosf(th), sn = sinf(th), ic = 1.f - c;
            float R00 = c + ic*ux*ux,     R01 = ic*ux*uy - sn*uz, R02 = ic*ux*uz + sn*uy;
            float R10 = ic*ux*uy + sn*uz, R11 = c + ic*uy*uy,     R12 = ic*uy*uz - sn*ux;
            float R20 = ic*ux*uz - sn*uy, R21 = ic*uy*uz + sn*ux, R22 = c + ic*uz*uz;

            float Jx = Jin[i*3+0]*s + tr0;
            float Jy = Jin[i*3+1]*s + tr1;
            float Jz = Jin[i*3+2]*s + tr2;
            int p = d_par[i];
            float tx, ty, tz;
            if (p < 0) { tx = Jx; ty = Jy; tz = Jz; }
            else {
                tx = Jx - (Jin[p*3+0]*s + tr0);
                ty = Jy - (Jin[p*3+1]*s + tr1);
                tz = Jz - (Jin[p*3+2]*s + tr2);
            }
            float G[12];
            if (p < 0) {
                G[0]=R00; G[1]=R01; G[2]=R02;  G[3]=tx;
                G[4]=R10; G[5]=R11; G[6]=R12;  G[7]=ty;
                G[8]=R20; G[9]=R21; G[10]=R22; G[11]=tz;
            } else {
                const float* Gp = GLS[lane][p];
                #pragma unroll
                for (int r = 0; r < 3; ++r) {
                    float g0 = Gp[r*4+0], g1 = Gp[r*4+1], g2 = Gp[r*4+2], g3 = Gp[r*4+3];
                    G[r*4+0] = g0*R00 + g1*R10 + g2*R20;
                    G[r*4+1] = g0*R01 + g1*R11 + g2*R21;
                    G[r*4+2] = g0*R02 + g1*R12 + g2*R22;
                    G[r*4+3] = g0*tx  + g1*ty  + g2*tz + g3;
                }
            }
            #pragma unroll
            for (int k = 0; k < 12; ++k) GLS[lane][i][k] = G[k];
            // A = G with translation column reduced by G_rot * J
            #pragma unroll
            for (int r = 0; r < 3; ++r) {
                float d = G[r*4+0]*Jx + G[r*4+1]*Jy + G[r*4+2]*Jz;
                ALS[lane][i][r*4+0] = G[r*4+0];
                ALS[lane][i][r*4+1] = G[r*4+1];
                ALS[lane][i][r*4+2] = G[r*4+2];
                ALS[lane][i][r*4+3] = G[r*4+3] - d;
            }
        }
    }
    __syncthreads();

    // invert star-pose affines (adjugate 3x3, then -Rinv*t)
    if (lane < NJ) {
        const float* M = ALS[16][lane];
        float r00=M[0], r01=M[1], r02=M[2],  t0=M[3];
        float r10=M[4], r11=M[5], r12=M[6],  t1=M[7];
        float r20=M[8], r21=M[9], r22=M[10], t2=M[11];
        float c00 = r11*r22 - r12*r21;
        float c01 = r12*r20 - r10*r22;
        float c02 = r10*r21 - r11*r20;
        float id  = 1.f / (r00*c00 + r01*c01 + r02*c02);
        float i00 = c00*id, i01 = (r02*r21 - r01*r22)*id, i02 = (r01*r12 - r02*r11)*id;
        float i10 = c01*id, i11 = (r00*r22 - r02*r20)*id, i12 = (r02*r10 - r00*r12)*id;
        float i20 = c02*id, i21 = (r01*r20 - r00*r21)*id, i22 = (r00*r11 - r01*r10)*id;
        AINV[lane][0]=i00; AINV[lane][1]=i01; AINV[lane][2]=i02;
        AINV[lane][3]= -(i00*t0 + i01*t1 + i02*t2);
        AINV[lane][4]=i10; AINV[lane][5]=i11; AINV[lane][6]=i12;
        AINV[lane][7]= -(i10*t0 + i11*t1 + i12*t2);
        AINV[lane][8]=i20; AINV[lane][9]=i21; AINV[lane][10]=i22;
        AINV[lane][11]= -(i20*t0 + i21*t1 + i22*t2);
    }
    __syncthreads();

    if (lane < B && lane < 16) {
        _Float16* tb = tab + lane * 1024;         // 32x32 f16 per batch
        for (int row = 0; row < 32; ++row) {
            for (int col = 0; col < 32; ++col) {
                float v = 0.f;
                if (col < NJ) {
                    if (row < 12)       v = AINV[col][row];
                    else if (row < 24)  v = ALS[lane][col][row-12];
                    else if (row == 24) v = 1.f;
                }
                tb[row*32 + col] = (_Float16)v;
            }
        }
    }
}

// ---------------------------------------------------------------------------
// Main: one wave per 32-vertex group.
// D[32rows x 32verts] = T[b] (32x32, A-frag) x Wt (32x32, B-frags) via 4 WMMAs,
// staged through per-wave LDS, then per-lane finalize (two 3x4 mat-vecs).
// ---------------------------------------------------------------------------
__global__ void __launch_bounds__(256)
skin_main(const float* __restrict__ verts,
          const float* __restrict__ wts,
          const _Float16* __restrict__ tab,
          float* __restrict__ out, int V, int ngroups)
{
    __shared__ float tile[8][32][33];             // per-wave D staging
    const int lane = threadIdx.x & 31;
    const int wave = threadIdx.x >> 5;
    const int b    = blockIdx.y;
    const int half = lane >> 4;                   // 0: lanes 0-15, 1: lanes 16-31
    const int l16  = lane & 15;

    // ---- A fragments (loaded once per wave; contiguous K layout matching B):
    // lanes 0-15: M=lane, elems 0..7 = K0..7,  elems 8..15 = K8..15
    // lanes16-31: M=lane-16, elems 0..7 = K16..23, elems 8..15 = K24..31
    const _Float16* tb = tab + (size_t)b * 1024;
    const int kA = half ? 16 : 0;
    const int kB = half ? 24 : 8;
    v16h a0, a1;
    {
        v8h lo = *(const v8h*)(tb + (l16     )*32 + kA);
        v8h hi = *(const v8h*)(tb + (l16     )*32 + kB);
        #pragma unroll
        for (int j = 0; j < 8; ++j) { a0[j] = lo[j]; a0[8+j] = hi[j]; }
        lo = *(const v8h*)(tb + (16 + l16)*32 + kA);
        hi = *(const v8h*)(tb + (16 + l16)*32 + kB);
        #pragma unroll
        for (int j = 0; j < 8; ++j) { a1[j] = lo[j]; a1[8+j] = hi[j]; }
    }

    const int gstride = gridDim.x * (blockDim.x >> 5);
    const size_t wbase = (size_t)b * V;

    for (int g = blockIdx.x * (blockDim.x >> 5) + wave; g < ngroups; g += gstride) {
        const int vbase = g * 32;

        // ---- B fragments (weights -> f16). Contiguous K layout, same order as A:
        // lanes 0-15: N=lane,   elems = K0..15
        // lanes16-31: N=lane-16, elems = K16..31 (K>=24 are zero)
        int col0 = vbase + l16;       if (col0 > V-1) col0 = V-1;
        int col1 = vbase + 16 + l16;  if (col1 > V-1) col1 = V-1;
        const v4f* w0 = (const v4f*)(wts + (wbase + (size_t)col0) * 24);
        const v4f* w1 = (const v4f*)(wts + (wbase + (size_t)col1) * 24);
        v16h bf0, bf1;
        if (half == 0) {
            v4f q0 = __builtin_nontemporal_load(w0 + 0);
            v4f q1 = __builtin_nontemporal_load(w0 + 1);
            v4f q2 = __builtin_nontemporal_load(w0 + 2);
            v4f q3 = __builtin_nontemporal_load(w0 + 3);
            v4f p0 = __builtin_nontemporal_load(w1 + 0);
            v4f p1 = __builtin_nontemporal_load(w1 + 1);
            v4f p2 = __builtin_nontemporal_load(w1 + 2);
            v4f p3 = __builtin_nontemporal_load(w1 + 3);
            #pragma unroll
            for (int j = 0; j < 4; ++j) {
                bf0[j]    = (_Float16)q0[j]; bf0[4+j]  = (_Float16)q1[j];
                bf0[8+j]  = (_Float16)q2[j]; bf0[12+j] = (_Float16)q3[j];
                bf1[j]    = (_Float16)p0[j]; bf1[4+j]  = (_Float16)p1[j];
                bf1[8+j]  = (_Float16)p2[j]; bf1[12+j] = (_Float16)p3[j];
            }
        } else {
            v4f q0 = __builtin_nontemporal_load(w0 + 4);
            v4f q1 = __builtin_nontemporal_load(w0 + 5);
            v4f p0 = __builtin_nontemporal_load(w1 + 4);
            v4f p1 = __builtin_nontemporal_load(w1 + 5);
            #pragma unroll
            for (int j = 0; j < 4; ++j) {
                bf0[j]   = (_Float16)q0[j]; bf0[4+j] = (_Float16)q1[j];
                bf1[j]   = (_Float16)p0[j]; bf1[4+j] = (_Float16)p1[j];
                bf0[8+j] = (_Float16)0.f;   bf0[12+j] = (_Float16)0.f;
                bf1[8+j] = (_Float16)0.f;   bf1[12+j] = (_Float16)0.f;
            }
        }

        v8f z = {};
        v8f d00 = __builtin_amdgcn_wmma_f32_16x16x32_f16(false, a0, false, bf0, (short)0, z, false, false);
        v8f d10 = __builtin_amdgcn_wmma_f32_16x16x32_f16(false, a1, false, bf0, (short)0, z, false, false);
        v8f d01 = __builtin_amdgcn_wmma_f32_16x16x32_f16(false, a0, false, bf1, (short)0, z, false, false);
        v8f d11 = __builtin_amdgcn_wmma_f32_16x16x32_f16(false, a1, false, bf1, (short)0, z, false, false);

        // ---- stage D to LDS (C/D layout: VGPR r -> M = r (lanes 0-15) / r+8)
        const int rb = half ? 8 : 0;
        #pragma unroll
        for (int r = 0; r < 8; ++r) {
            tile[wave][rb + r     ][l16     ] = d00[r];
            tile[wave][rb + r + 16][l16     ] = d10[r];
            tile[wave][rb + r     ][l16 + 16] = d01[r];
            tile[wave][rb + r + 16][l16 + 16] = d11[r];
        }
        __asm__ volatile("s_wait_dscnt 0" ::: "memory");  // per-wave LDS in-order; fence compiler + HW

        // ---- finalize: lane -> vertex vbase+lane, column = lane
        float d[25];
        #pragma unroll
        for (int r = 0; r < 25; ++r) d[r] = tile[wave][r][lane];

        const int v = vbase + lane;
        const int vc = v > V-1 ? V-1 : v;
        const float* vp = verts + (wbase + (size_t)vc) * 3;
        float x = vp[0], y = vp[1], zz = vp[2];
        float tx = d[0]*x + d[1]*y + d[2]*zz  + d[3];
        float ty = d[4]*x + d[5]*y + d[6]*zz  + d[7];
        float tz = d[8]*x + d[9]*y + d[10]*zz + d[11];
        float W  = d[24];
        float ox = d[12]*tx + d[13]*ty + d[14]*tz + d[15]*W;
        float oy = d[16]*tx + d[17]*ty + d[18]*tz + d[19]*W;
        float oz = d[20]*tx + d[21]*ty + d[22]*tz + d[23]*W;
        if (v < V) {
            float* op = out + (wbase + (size_t)v) * 3;
            __builtin_nontemporal_store(ox, op + 0);
            __builtin_nontemporal_store(oy, op + 1);
            __builtin_nontemporal_store(oz, op + 2);
        }
        __asm__ volatile("" ::: "memory");                // don't sink next iter's DS stores above reads
    }
}

extern "C" void kernel_launch(void* const* d_in, const int* in_sizes, int n_in,
                              void* d_out, int out_size, void* d_ws, size_t ws_size,
                              hipStream_t stream) {
    (void)n_in; (void)out_size; (void)ws_size;
    const float* verts = (const float*)d_in[0];
    const float* pose  = (const float*)d_in[1];
    const float* wts   = (const float*)d_in[2];
    const float* scale = (const float*)d_in[3];
    const float* trans = (const float*)d_in[4];
    const float* J     = (const float*)d_in[5];
    float* out = (float*)d_out;

    const int B = in_sizes[1] / 72;                 // pose [B,72]
    const int V = in_sizes[0] / (3 * B);            // vertices [B,V,3]
    _Float16* tab = (_Float16*)d_ws;                // B * 32*32 f16 = 32 KB

    skin_prep<<<1, 32, 0, stream>>>(pose, scale, trans, J, tab, B);

    const int ngroups = (V + 31) / 32;
    const int waves_per_block = 256 / 32;
    const int gx = (ngroups + waves_per_block - 1) / waves_per_block;
    dim3 grid(gx, B, 1);
    skin_main<<<grid, 256, 0, stream>>>(verts, wts, tab, out, V, ngroups);
}
